// LanguageModel_78726750536271
// MI455X (gfx1250) — matrix operand order repaired and checked by reading, not compiled, not run on hardware
//
#include <hip/hip_runtime.h>
#include <stdint.h>
#include <math.h>

// ---------------- problem constants ----------------
#define VOCAB  10000
#define BSZ    256
#define TLEN   64
#define RNN    512
#define VIS    1024
#define WORD   512
#define NOUT   10001      // VOCAB + 1
#define NOUTP  10048      // padded to multiple of 64 for 32x64 wave tiles

typedef __attribute__((ext_vector_type(16))) __bf16 v16bf;
typedef __attribute__((ext_vector_type(8)))  float  v8f;

union frag16 { v16bf v; uint32_t u[8]; uint4 q[2]; };

__device__ __forceinline__ uint16_t f2bf(float f) {
  union { float f; uint32_t u; } x; x.f = f;
  uint32_t r = x.u + 0x7FFFu + ((x.u >> 16) & 1u);   // round-to-nearest-even
  return (uint16_t)(r >> 16);
}
__device__ __forceinline__ float sigmoidf_(float x) {
  return 1.0f / (1.0f + __expf(-x));
}

// ---------------- weight conversion / packing kernels ----------------
__global__ void k_cast_f32_bf16(const float* __restrict__ s, uint16_t* __restrict__ d, int n) {
  int i = blockIdx.x * blockDim.x + threadIdx.x;
  if (i < n) d[i] = f2bf(s[i]);
}

// Pack a row-major f32 [Krows, Nsrc] weight into WMMA wave32 B-fragment blocks
// (zero-padded to Ndst columns). dst dword index:
//   (( (k/32) * (Ndst/16) + (n/16) ) * 32 + lane) * 8 + v
// where lane = 16*(K-half) + n%16, and dword v holds the bf16 pair (K= kc*32 + half*16 + 2v, +1).
__global__ void k_pack_wmmaB(const float* __restrict__ s, uint32_t* __restrict__ d,
                             int Krows, int Nsrc, int Ndst) {
  int i = blockIdx.x * blockDim.x + threadIdx.x;
  int total = (Krows >> 5) * (Ndst >> 4) * 256;
  if (i >= total) return;
  int v    = i & 7;
  int lane = (i >> 3) & 31;
  int blk  = i >> 8;
  int ntiles = Ndst >> 4;
  int kc = blk / ntiles, nt = blk - kc * ntiles;
  int n  = nt * 16 + (lane & 15);
  int k0 = kc * 32 + (lane >> 4) * 16 + 2 * v;
  uint32_t lo = 0, hi = 0;
  if (n < Nsrc) {
    lo = f2bf(s[(size_t)k0 * Nsrc + n]);
    hi = f2bf(s[(size_t)(k0 + 1) * Nsrc + n]);
  }
  d[i] = lo | (hi << 16);
}

__global__ void k_pad_bias(const float* __restrict__ s, float* __restrict__ d, int Nsrc, int Ndst) {
  int i = blockIdx.x * blockDim.x + threadIdx.x;
  if (i < Ndst) d[i] = (i < Nsrc) ? s[i] : 0.0f;
}

// zero h-state (bf16 half of X2) and c-state (f32)
__global__ void k_init_state(uint16_t* __restrict__ X2, float* __restrict__ c) {
  int i = blockIdx.x * blockDim.x + threadIdx.x;     // BSZ*RNN threads
  if (i >= BSZ * RNN) return;
  int b = i >> 9, j = i & 511;
  X2[b * (WORD + RNN) + WORD + j] = 0;
  c[i] = 0.0f;
}

// ---------------- bf16 WMMA GEMM: C[M,N] = A[M,K]*Bpk (+base)(+bias) ----------------
// A row-major bf16 [M,K] (lda). Bpk pre-packed fragment blocks. Each wave: 32(M) x 64(N).
__global__ __launch_bounds__(256)
void k_gemm_bf16_wmma(const uint16_t* __restrict__ A, const uint32_t* __restrict__ Bpk,
                      const float* __restrict__ Cbase, const float* __restrict__ bias,
                      float* __restrict__ C,
                      int M, int N, int K, int lda, int ldc) {
  const int wave = threadIdx.x >> 5;
  const int lane = threadIdx.x & 31;
  const int tilesN  = N >> 6;                    // 64-col tiles
  const int ntilesN = N >> 4;                    // 16-col fragment tiles
  const int total   = tilesN * (M >> 5);         // 32-row tiles
  const int tile    = blockIdx.x * (blockDim.x >> 5) + wave;
  if (tile >= total) return;                     // wave-uniform: EXEC stays all-ones
  const int n0   = (tile % tilesN) << 6;
  const int m0   = (tile / tilesN) << 5;
  const int row  = lane & 15;
  const int half = lane >> 4;

  v8f acc[2][4] = {};

  const uint16_t* arow0 = A + (size_t)(m0 + row) * lda      + half * 16;
  const uint16_t* arow1 = A + (size_t)(m0 + 16 + row) * lda + half * 16;
  const uint32_t* bbase = Bpk + (size_t)(n0 >> 4) * 256 + lane * 8;
  const size_t    bstep = (size_t)ntilesN * 256;             // dwords per K-chunk

  for (int k = 0; k < K; k += 32) {
    // ---- two A fragments (rows m0..m0+15 and m0+16..m0+31): 2x b128 each ----
    frag16 a0, a1;
    a0.q[0] = *(const uint4*)(arow0 + k);
    a0.q[1] = *(const uint4*)(arow0 + k + 8);
    a1.q[0] = *(const uint4*)(arow1 + k);
    a1.q[1] = *(const uint4*)(arow1 + k + 8);

    // ---- four B fragments (cols n0 .. n0+63): 2x b128 each, linear packed ----
    const uint32_t* bb = bbase + (size_t)(k >> 5) * bstep;
    frag16 b0, b1, b2, b3;
    b0.q[0] = *(const uint4*)(bb);        b0.q[1] = *(const uint4*)(bb + 4);
    b1.q[0] = *(const uint4*)(bb + 256);  b1.q[1] = *(const uint4*)(bb + 260);
    b2.q[0] = *(const uint4*)(bb + 512);  b2.q[1] = *(const uint4*)(bb + 516);
    b3.q[0] = *(const uint4*)(bb + 768);  b3.q[1] = *(const uint4*)(bb + 772);

    if (k + 32 < K) {
      __builtin_prefetch(bb + bstep, 0, 0);                  // global_prefetch_b8
      __builtin_prefetch(arow0 + k + 32, 0, 0);
    }

    acc[0][0] = __builtin_amdgcn_wmma_f32_16x16x32_bf16(false, a0.v, false, b0.v, (short)0, acc[0][0], false, false);
    acc[0][1] = __builtin_amdgcn_wmma_f32_16x16x32_bf16(false, a0.v, false, b1.v, (short)0, acc[0][1], false, false);
    acc[0][2] = __builtin_amdgcn_wmma_f32_16x16x32_bf16(false, a0.v, false, b2.v, (short)0, acc[0][2], false, false);
    acc[0][3] = __builtin_amdgcn_wmma_f32_16x16x32_bf16(false, a0.v, false, b3.v, (short)0, acc[0][3], false, false);
    acc[1][0] = __builtin_amdgcn_wmma_f32_16x16x32_bf16(false, a1.v, false, b0.v, (short)0, acc[1][0], false, false);
    acc[1][1] = __builtin_amdgcn_wmma_f32_16x16x32_bf16(false, a1.v, false, b1.v, (short)0, acc[1][1], false, false);
    acc[1][2] = __builtin_amdgcn_wmma_f32_16x16x32_bf16(false, a1.v, false, b2.v, (short)0, acc[1][2], false, false);
    acc[1][3] = __builtin_amdgcn_wmma_f32_16x16x32_bf16(false, a1.v, false, b3.v, (short)0, acc[1][3], false, false);
  }

  // ---- epilogue: C layout lane N=lane%16, VGPR v -> M = v + 8*half ----
#pragma unroll
  for (int mt = 0; mt < 2; ++mt) {
#pragma unroll
    for (int nt = 0; nt < 4; ++nt) {
#pragma unroll
      for (int v = 0; v < 8; ++v) {
        const int mr = m0 + mt * 16 + v + half * 8;
        const int nc = n0 + nt * 16 + row;
        float val = acc[mt][nt][v];
        if (Cbase) val += Cbase[(size_t)mr * ldc + nc];
        if (bias)  val += bias[nc];
        C[(size_t)mr * ldc + nc] = val;
      }
    }
  }
}

// ---------------- embedding gather -> bf16 word half of X2 ----------------
__global__ void k_embed(const int* __restrict__ seqz, const float* __restrict__ emb,
                        uint16_t* __restrict__ X2, int t) {
  int i = blockIdx.x * blockDim.x + threadIdx.x;     // BSZ*WORD
  if (i >= BSZ * WORD) return;
  int b = i >> 9, j = i & 511;
  int widx = (t == 0) ? (VOCAB + 1) : seqz[b * TLEN + t - 1];
  X2[b * (WORD + RNN) + j] = f2bf(emb[(size_t)widx * WORD + j]);
}

// ---------------- LSTM + sentinel pointwise ----------------
__global__ void k_lstm_pointwise(const float* __restrict__ gates,
                                 const float* __restrict__ glin,
                                 float* __restrict__ c,
                                 uint16_t* __restrict__ X2,
                                 uint16_t* __restrict__ sbf) {
  int i = blockIdx.x * blockDim.x + threadIdx.x;     // BSZ*RNN
  if (i >= BSZ * RNN) return;
  int b = i >> 9, j = i & 511;
  const float* g = gates + (size_t)b * (4 * RNN);
  float ig = sigmoidf_(g[j]);
  float fg = sigmoidf_(g[RNN + j]);
  float gg = tanhf(g[2 * RNN + j]);
  float og = sigmoidf_(g[3 * RNN + j]);
  float cn = fg * c[i] + ig * gg;
  float tc = tanhf(cn);
  float hn = og * tc;
  float sg = sigmoidf_(glin[i]);
  c[i] = cn;
  X2[b * (WORD + RNN) + WORD + j] = f2bf(hn);        // h_t for next step (bf16)
  sbf[i] = f2bf(sg * tc);                            // sentinel-gated s_t
}

// ---------------- log_softmax over 10001 + target gather, masked ----------------
__global__ __launch_bounds__(256)
void k_logsoftmax_gather(const float* __restrict__ logits, const int* __restrict__ seqz,
                         float* __restrict__ out, int t) {
  __shared__ float red[256];
  int b = blockIdx.x, tid = threadIdx.x;
  const float* lr = logits + (size_t)b * NOUTP;

  float mx = -3.4e38f;
  for (int j = tid; j < NOUT; j += 256) mx = fmaxf(mx, lr[j]);
  red[tid] = mx; __syncthreads();
  for (int s = 128; s > 0; s >>= 1) {
    if (tid < s) red[tid] = fmaxf(red[tid], red[tid + s]);
    __syncthreads();
  }
  mx = red[0]; __syncthreads();

  float sum = 0.0f;
  for (int j = tid; j < NOUT; j += 256) sum += __expf(lr[j] - mx);
  red[tid] = sum; __syncthreads();
  for (int s = 128; s > 0; s >>= 1) {
    if (tid < s) red[tid] += red[tid + s];
    __syncthreads();
  }

  if (tid == 0) {
    int tgt = seqz[b * TLEN + t];
    float m = (t == 0) ? 1.0f : ((seqz[b * TLEN + t - 1] != 0) ? 1.0f : 0.0f);
    out[t * BSZ + b] = m * (lr[tgt] - mx - logf(red[0]));
  }
}

// ---------------- host orchestration ----------------
extern "C" void kernel_launch(void* const* d_in, const int* in_sizes, int n_in,
                              void* d_out, int out_size, void* d_ws, size_t ws_size,
                              hipStream_t stream) {
  const float* vis    = (const float*)d_in[0];
  const int*   seqz   = (const int*)  d_in[1];
  const float* emb    = (const float*)d_in[2];
  const float* W_ih   = (const float*)d_in[3];
  const float* W_hh   = (const float*)d_in[4];
  const float* b_lstm = (const float*)d_in[5];
  const float* vis2gW = (const float*)d_in[6];
  const float* vis2gb = (const float*)d_in[7];
  const float* w2gW   = (const float*)d_in[8];
  const float* h2gW   = (const float*)d_in[9];
  const float* outW   = (const float*)d_in[10];
  const float* outb   = (const float*)d_in[11];
  float* out = (float*)d_out;

  // packed B sizes (dwords): (K/32) * (N/16) * 256
  auto packDw = [](int K, int N) { return (size_t)(K >> 5) * (N >> 4) * 256; };

  // workspace carve-out (256B aligned)
  uint8_t* p = (uint8_t*)d_ws;
  auto carve = [&](size_t bytes) { uint8_t* r = p; p += (bytes + 255) & ~(size_t)255; return r; };
  uint32_t* W2pk   = (uint32_t*)carve(packDw(1024, 2048) * 4);  // [W_ih_word ; W_hh]  packed
  uint32_t* Wvispk = (uint32_t*)carve(packDw(1024, 2048) * 4);  // W_ih vis rows       packed
  uint32_t* Wg2pk  = (uint32_t*)carve(packDw(1024, 512)  * 4);  // [w2g_W ; h2g_W]     packed
  uint32_t* Wv2gpk = (uint32_t*)carve(packDw(1024, 512)  * 4);  // vis2g_W             packed
  uint32_t* Woutpk = (uint32_t*)carve(packDw(512, NOUTP) * 4);  // out_W padded        packed
  uint16_t* visbf  = (uint16_t*)carve((size_t)BSZ * VIS * 2);   // vis_feats bf16 (A-side)
  uint16_t* X2     = (uint16_t*)carve((size_t)BSZ * (WORD + RNN) * 2); // [xw | h] bf16
  uint16_t* sbf    = (uint16_t*)carve((size_t)BSZ * RNN * 2);   // s_t bf16
  float*    gbase  = (float*)carve((size_t)BSZ * 2048 * 4);     // vis @ W_ih_vis + b
  float*    ggbase = (float*)carve((size_t)BSZ * 512  * 4);     // vis @ vis2g + b
  float*    gates  = (float*)carve((size_t)BSZ * 2048 * 4);
  float*    glin   = (float*)carve((size_t)BSZ * 512  * 4);
  float*    cstate = (float*)carve((size_t)BSZ * 512  * 4);
  float*    logits = (float*)carve((size_t)BSZ * NOUTP * 4);
  float*    outbP  = (float*)carve((size_t)NOUTP * 4);

  const int TPB = 256;
  auto pack = [&](const float* s, uint32_t* d, int Krows, int Nsrc, int Ndst) {
    int total = (Krows >> 5) * (Ndst >> 4) * 256;
    k_pack_wmmaB<<<(total + TPB - 1) / TPB, TPB, 0, stream>>>(s, d, Krows, Nsrc, Ndst);
  };
  auto gemm = [&](const uint16_t* A, const uint32_t* Bpk, const float* base, const float* bias,
                  float* C, int M, int N, int K, int lda, int ldc) {
    int total  = (N / 64) * (M / 32);
    int blocks = (total + 7) / 8;                  // 8 waves per 256-thread block
    k_gemm_bf16_wmma<<<blocks, TPB, 0, stream>>>(A, Bpk, base, bias, C, M, N, K, lda, ldc);
  };

  // ---- one-time weight packing into WMMA fragment layout ----
  pack(W_ih + (size_t)1024 * 2048, W2pk,                     512, 2048, 2048); // word rows (K 0..511)
  pack(W_hh,                       W2pk + packDw(512, 2048), 512, 2048, 2048); // stacked W_hh (K 512..1023)
  pack(W_ih,                       Wvispk,                  1024, 2048, 2048); // vis rows
  pack(w2gW,                       Wg2pk,                    512,  512,  512);
  pack(h2gW,                       Wg2pk + packDw(512, 512), 512,  512,  512);
  pack(vis2gW,                     Wv2gpk,                  1024,  512,  512);
  pack(outW,                       Woutpk,                   512, NOUT, NOUTP);
  k_cast_f32_bf16<<<(BSZ * VIS + TPB - 1) / TPB, TPB, 0, stream>>>(vis, visbf, BSZ * VIS);
  k_pad_bias<<<(NOUTP + TPB - 1) / TPB, TPB, 0, stream>>>(outb, outbP, NOUT, NOUTP);

  // ---- time-invariant bases ----
  gemm(visbf, Wvispk, nullptr, b_lstm, gbase, BSZ, 2048, 1024, 1024, 2048);
  gemm(visbf, Wv2gpk, nullptr, vis2gb, ggbase, BSZ, 512, 1024, 1024, 512);

  // ---- zero h, c ----
  k_init_state<<<(BSZ * RNN + TPB - 1) / TPB, TPB, 0, stream>>>(X2, cstate);

  // ---- sequential decode ----
  for (int t = 0; t < TLEN; ++t) {
    k_embed<<<(BSZ * WORD + TPB - 1) / TPB, TPB, 0, stream>>>(seqz, emb, X2, t);
    // gates = [xw|h] @ [W_ih_word; W_hh] + (vis@W_ih_vis + b_lstm)
    gemm(X2, W2pk, gbase, nullptr, gates, BSZ, 2048, 1024, 1024, 2048);
    // glin = [xw|h] @ [w2g; h2g] + (vis@vis2g + b)
    gemm(X2, Wg2pk, ggbase, nullptr, glin, BSZ, 512, 1024, 1024, 512);
    k_lstm_pointwise<<<(BSZ * RNN + TPB - 1) / TPB, TPB, 0, stream>>>(gates, glin, cstate, X2, sbf);
    // logits = s_t @ out_W + out_b
    gemm(sbf, Woutpk, nullptr, outbP, logits, BSZ, NOUTP, 512, 512, NOUTP);
    k_logsoftmax_gather<<<BSZ, TPB, 0, stream>>>(logits, seqz, out, t);
  }
}
